// GATRatingPrediction_15315853377884
// MI455X (gfx1250) — matrix operand order repaired and checked by reading, not compiled, not run on hardware
//
#include <hip/hip_runtime.h>
#include <hip/hip_bf16.h>

// ---------------------------------------------------------------------------
// GAT rating prediction for gfx1250 (MI455X), wave32 + WMMA bf16.
// Pipeline per layer:
//   1) gemm_wmma<K>   : xw = X @ W            (v_wmma_f32_16x16x32_bf16, K unrolled)
//   2) row_logits     : s_src/s_dst = xw @ a  (wave-per-node shfl reduce)
//   3) fills          : mkey=0, denom=0, out=0
//   4) edge_logit_max : e = lrelu(...), segment-max via u32-key atomicMax
//   5) edge_exp       : ex = exp(e - m[dst]), segment-sum via atomicAdd
//   6) edge_agg       : out[dst] += (ex/denom[dst]) * xw[src]  (wave/edge)
//   7) bias_act       : out += b (+ ReLU for layer 1)
// Head: wave-per-sample dot of [h[u], h[m]] with fc_W.
// ---------------------------------------------------------------------------

typedef __attribute__((ext_vector_type(16))) __bf16 v16bf;
typedef __attribute__((ext_vector_type(8)))  float  v8f;

#define GAT_H 64
#define NEG_SLOPE 0.2f

// ---------------- utility ----------------

__global__ void fill_u32_kernel(unsigned* __restrict__ p, unsigned v, int n) {
  int i = blockIdx.x * blockDim.x + threadIdx.x;
  if (i < n) p[i] = v;
}

__device__ __forceinline__ unsigned fkey(float f) {
  unsigned u = __float_as_uint(f);
  return (u & 0x80000000u) ? ~u : (u | 0x80000000u);
}
__device__ __forceinline__ float funkey(unsigned k) {
  unsigned u = (k & 0x80000000u) ? (k & 0x7FFFFFFFu) : ~k;
  return __uint_as_float(u);
}

__device__ __forceinline__ void get_edge(const int* __restrict__ ei, int E, int i,
                                         int& s, int& d) {
  if (i < E) { s = ei[i]; d = ei[E + i]; }
  else       { s = d = i - E; }          // self-loop edges appended after E
}

// ---------------- 1) GEMM via WMMA bf16: C[N,64] = A[N,K] @ W[K,64] ----------
// K is compile-time (256 for layer 1, 64 for layer 2) -> fully unrolled WMMA chain.

template <int K>
__global__ __launch_bounds__(256)
void gemm_wmma_kernel(const float* __restrict__ A,
                      const float* __restrict__ W,
                      float* __restrict__ C, int N) {
  const int lane = threadIdx.x & 31;
  const int wave = threadIdx.x >> 5;              // 0..7
  const int rowTile = blockIdx.x * 2 + (wave >> 2);
  const int colTile = wave & 3;                   // 4 tiles of 16 cols = 64
  const int row0 = rowTile * 16;
  const int col0 = colTile * 16;
  if (row0 >= N) return;                          // wave-uniform

  const int hi = lane >> 4;                       // lane half (0/1)
  const int ln = lane & 15;
  const int arow = row0 + ln;                     // A row for this lane
  const int bcol = col0 + ln;                     // W col for this lane
  const float* __restrict__ arowp = A + (size_t)arow * K;

  v8f acc = {};
#pragma unroll
  for (int k0 = 0; k0 < K; k0 += 32) {
    // A tile (16x32 bf16): lanes 0-15 K=[0..7,16..23], lanes 16-31 K=[8..15,24..31]
    const float4* ap0 = reinterpret_cast<const float4*>(arowp + k0 + hi * 8);
    const float4* ap1 = reinterpret_cast<const float4*>(arowp + k0 + 16 + hi * 8);
    float4 f0 = ap0[0], f1 = ap0[1];   // K = base .. base+7
    float4 f2 = ap1[0], f3 = ap1[1];   // K = base+16 .. base+23
    v16bf a;
    a[0]  = (__bf16)f0.x; a[1]  = (__bf16)f0.y; a[2]  = (__bf16)f0.z; a[3]  = (__bf16)f0.w;
    a[4]  = (__bf16)f1.x; a[5]  = (__bf16)f1.y; a[6]  = (__bf16)f1.z; a[7]  = (__bf16)f1.w;
    a[8]  = (__bf16)f2.x; a[9]  = (__bf16)f2.y; a[10] = (__bf16)f2.z; a[11] = (__bf16)f2.w;
    a[12] = (__bf16)f3.x; a[13] = (__bf16)f3.y; a[14] = (__bf16)f3.z; a[15] = (__bf16)f3.w;
    // B tile (32x16 bf16): lane = column, lanes 0-15 K=[0..7,16..23], lanes 16-31 K=[8..15,24..31]
    v16bf b;
#pragma unroll
    for (int e = 0; e < 16; ++e)
      b[e] = (__bf16)W[(size_t)(k0 + hi * 8 + (e & 7) + (e >> 3) * 16) * GAT_H + bcol];

    acc = __builtin_amdgcn_wmma_f32_16x16x32_bf16(
        /*neg_a=*/false, a, /*neg_b=*/false, b,
        /*c_mod=*/(short)0, acc, /*reuse_a=*/false, /*reuse_b=*/false);
  }

  // C/D layout: VGPR r -> row (row0 + 8*hi + r), col = col0 + ln
  if (row0 + 16 <= N) {                 // full tile: unguarded strided stores
    float* crow = C + (size_t)(row0 + 8 * hi) * GAT_H + col0 + ln;
#pragma unroll
    for (int r = 0; r < 8; ++r) crow[(size_t)r * GAT_H] = acc[r];
  } else {                              // ragged tail (not hit when N % 16 == 0)
#pragma unroll
    for (int r = 0; r < 8; ++r) {
      int rr = row0 + 8 * hi + r;
      if (rr < N) C[(size_t)rr * GAT_H + col0 + ln] = acc[r];
    }
  }
}

// ---------------- 2) per-node attention logits ----------------

__global__ void row_logits_kernel(const float* __restrict__ xw,
                                  const float* __restrict__ a_src,
                                  const float* __restrict__ a_dst,
                                  float* __restrict__ s_src,
                                  float* __restrict__ s_dst, int N) {
  const int lane = threadIdx.x & 31;
  const int node = blockIdx.x * 8 + (threadIdx.x >> 5);
  if (node >= N) return;
  const float* r = xw + (size_t)node * GAT_H;
  float v0 = r[lane], v1 = r[lane + 32];
  float ss = v0 * a_src[lane] + v1 * a_src[lane + 32];
  float sd = v0 * a_dst[lane] + v1 * a_dst[lane + 32];
#pragma unroll
  for (int o = 16; o > 0; o >>= 1) {
    ss += __shfl_xor(ss, o, 32);
    sd += __shfl_xor(sd, o, 32);
  }
  if (lane == 0) { s_src[node] = ss; s_dst[node] = sd; }
}

// ---------------- 4) edge logits + segment max ----------------

__global__ void edge_logit_max_kernel(const int* __restrict__ ei, int E, int T,
                                      const float* __restrict__ s_src,
                                      const float* __restrict__ s_dst,
                                      float* __restrict__ ebuf,
                                      unsigned* __restrict__ mkey) {
  int i = blockIdx.x * blockDim.x + threadIdx.x;
  if (i >= T) return;
  int s, d;
  get_edge(ei, E, i, s, d);
  float e = s_src[s] + s_dst[d];
  e = (e > 0.0f) ? e : NEG_SLOPE * e;   // leaky_relu
  ebuf[i] = e;
  atomicMax(&mkey[d], fkey(e));
}

// ---------------- 5) exp + segment sum ----------------

__global__ void edge_exp_kernel(const int* __restrict__ ei, int E, int T,
                                float* __restrict__ ebuf,
                                const unsigned* __restrict__ mkey,
                                float* __restrict__ denom) {
  int i = blockIdx.x * blockDim.x + threadIdx.x;
  if (i >= T) return;
  int s, d;
  get_edge(ei, E, i, s, d);
  float ex = __expf(ebuf[i] - funkey(mkey[d]));
  ebuf[i] = ex;
  atomicAdd(&denom[d], ex);
}

// ---------------- 6) weighted aggregation (wave per edge) ----------------

__global__ void edge_agg_kernel(const int* __restrict__ ei, int E, int T,
                                const float* __restrict__ ebuf,
                                const float* __restrict__ denom,
                                const float* __restrict__ xw,
                                float* __restrict__ outb) {
  const int lane = threadIdx.x & 31;
  const int e = blockIdx.x * 8 + (threadIdx.x >> 5);
  if (e >= T) return;
  int s, d;
  get_edge(ei, E, e, s, d);
  float alpha = ebuf[e] / denom[d];
  const float* xr = xw + (size_t)s * GAT_H;
  float* orow = outb + (size_t)d * GAT_H;
  atomicAdd(&orow[lane], alpha * xr[lane]);
  atomicAdd(&orow[lane + 32], alpha * xr[lane + 32]);
}

// ---------------- 7) bias (+ optional ReLU) ----------------

__global__ void bias_act_kernel(float* __restrict__ h, const float* __restrict__ b,
                                int total, int relu) {
  int i = blockIdx.x * blockDim.x + threadIdx.x;
  if (i >= total) return;
  float v = h[i] + b[i & (GAT_H - 1)];
  if (relu) v = fmaxf(v, 0.0f);
  h[i] = v;
}

// ---------------- head: concat-gather + FC ----------------

__global__ void fc_kernel(const float* __restrict__ h,
                          const int* __restrict__ user,
                          const int* __restrict__ movie,
                          const float* __restrict__ fcW,
                          const float* __restrict__ fcb,
                          float* __restrict__ out, int B) {
  const int lane = threadIdx.x & 31;
  const int b = blockIdx.x * 8 + (threadIdx.x >> 5);
  if (b >= B) return;
  const int u = user[b], m = movie[b];
  float s = 0.0f;
#pragma unroll
  for (int j = 0; j < 4; ++j) {
    int idx = lane + j * 32;   // 0..127
    float c = (idx < GAT_H) ? h[(size_t)u * GAT_H + idx]
                            : h[(size_t)m * GAT_H + (idx - GAT_H)];
    s += c * fcW[idx];
  }
#pragma unroll
  for (int o = 16; o > 0; o >>= 1) s += __shfl_xor(s, o, 32);
  if (lane == 0) out[b] = s + fcb[0];
}

// ---------------- host orchestration ----------------

extern "C" void kernel_launch(void* const* d_in, const int* in_sizes, int n_in,
                              void* d_out, int out_size, void* d_ws, size_t ws_size,
                              hipStream_t stream) {
  const float* x      = (const float*)d_in[0];
  const int*   ei     = (const int*)d_in[1];
  const int*   user   = (const int*)d_in[2];
  const int*   movie  = (const int*)d_in[3];
  const float* W1     = (const float*)d_in[4];
  const float* a_src1 = (const float*)d_in[5];
  const float* a_dst1 = (const float*)d_in[6];
  const float* b1     = (const float*)d_in[7];
  const float* W2     = (const float*)d_in[8];
  const float* a_src2 = (const float*)d_in[9];
  const float* a_dst2 = (const float*)d_in[10];
  const float* b2     = (const float*)d_in[11];
  const float* fcW    = (const float*)d_in[12];
  const float* fcb    = (const float*)d_in[13];
  float* out = (float*)d_out;

  const int FIN = in_sizes[4] / GAT_H;   // 256
  const int N   = in_sizes[0] / FIN;     // 100000
  const int E   = in_sizes[1] / 2;       // 1000000
  const int B   = in_sizes[2];           // 16384
  const int T   = E + N;                 // edges incl. self-loops

  // carve scratch from d_ws
  char* ws = (char*)d_ws;
  size_t off = 0;
  auto carve = [&](size_t bytes) -> void* {
    void* p = ws + off;
    off += (bytes + 255) & ~((size_t)255);
    return p;
  };
  float*    xw    = (float*)carve((size_t)N * GAT_H * 4);
  float*    hbuf  = (float*)carve((size_t)N * GAT_H * 4);
  float*    s_src = (float*)carve((size_t)N * 4);
  float*    s_dst = (float*)carve((size_t)N * 4);
  unsigned* mkey  = (unsigned*)carve((size_t)N * 4);
  float*    denom = (float*)carve((size_t)N * 4);
  float*    ebuf  = (float*)carve((size_t)T * 4);
  (void)ws_size; (void)n_in; (void)out_size;

  const int NH = N * GAT_H;

  auto layer_rest = [&](const float* as_, const float* ad_, const float* bias, int relu) {
    row_logits_kernel<<<dim3((N + 7) / 8), dim3(256), 0, stream>>>(xw, as_, ad_, s_src, s_dst, N);
    fill_u32_kernel<<<dim3((N + 255) / 256), dim3(256), 0, stream>>>(mkey, 0u, N);
    fill_u32_kernel<<<dim3((N + 255) / 256), dim3(256), 0, stream>>>((unsigned*)denom, 0u, N);
    fill_u32_kernel<<<dim3((NH + 255) / 256), dim3(256), 0, stream>>>((unsigned*)hbuf, 0u, NH);
    edge_logit_max_kernel<<<dim3((T + 255) / 256), dim3(256), 0, stream>>>(ei, E, T, s_src, s_dst, ebuf, mkey);
    edge_exp_kernel<<<dim3((T + 255) / 256), dim3(256), 0, stream>>>(ei, E, T, ebuf, mkey, denom);
    edge_agg_kernel<<<dim3((T + 7) / 8), dim3(256), 0, stream>>>(ei, E, T, ebuf, denom, xw, hbuf);
    bias_act_kernel<<<dim3((NH + 255) / 256), dim3(256), 0, stream>>>(hbuf, bias, NH, relu);
  };

  // Layer 1: K = 256 (compile-time specialization)
  gemm_wmma_kernel<256><<<dim3((N + 31) / 32), dim3(256), 0, stream>>>(x, W1, xw, N);
  layer_rest(a_src1, a_dst1, b1, /*relu=*/1);

  // Layer 2: K = 64
  gemm_wmma_kernel<64><<<dim3((N + 31) / 32), dim3(256), 0, stream>>>(hbuf, W2, xw, N);
  layer_rest(a_src2, a_dst2, b2, /*relu=*/0);

  fc_kernel<<<dim3((B + 7) / 8), dim3(256), 0, stream>>>(hbuf, user, movie, fcW, fcb, out, B);
}
